// get_model_49529562857490
// MI455X (gfx1250) — matrix operand order repaired and checked by reading, not compiled, not run on hardware
//
#include <hip/hip_runtime.h>
#include <math.h>

typedef __attribute__((ext_vector_type(16))) _Float16 v16h;
typedef __attribute__((ext_vector_type(8)))  float    v8f;

#define TPB 256

// ============================================================================
// WMMA GEMM, fully unguarded memory path.
//   Y[r, c] = act( acc[r,c] * scale[c] + shift[c] ),  acc = X @ Wp^T
//   X : [R, Kp]  row-major fp32, R multiple of 128, row stride Kp (mult of 32)
//   Wp: [CoP, Kp] packed fp16 weights, zero padded (CoP mult of 64)
//   scale/shift: [CoP], zero padded (makes pad columns exactly 0)
//   ldY = pad32(Co) = stride of Y rows; stores masked to gc < ldY.
// Block: 128x64 tile, 8 waves (4x2), each wave 2x2 WMMA 16x16 tiles.
// Staging: A loads batched into regs then converted (MLP-friendly);
//          B tile is a straight fp16 uint4 copy global->LDS.
// ============================================================================
__global__ __launch_bounds__(256) void gemm2_kernel(
    const float* __restrict__ X, const _Float16* __restrict__ Wp,
    const float* __restrict__ sc, const float* __restrict__ sh,
    float* __restrict__ Y, int Kp, int ldY, int relu)
{
    __shared__ _Float16 As[128][40];
    __shared__ _Float16 Bs[64][40];
    const int tid  = threadIdx.x;
    const int lane = tid & 31;
    const int wave = tid >> 5;
    const int wm   = wave >> 1;          // 0..3  -> 32-row slice
    const int wn   = wave & 1;           // 0..1  -> 32-col slice
    const int m0   = blockIdx.x * 128;
    const int n0   = blockIdx.y * 64;
    const int hsel = (lane >> 4) & 1;
    const int l15  = lane & 15;
    const int bn   = tid >> 2;           // 0..63  B row
    const int bkh  = (tid & 3) * 8;      // 0,8,16,24  B k-offset (halves)

    v8f acc[2][2] = {};
    for (int k0 = 0; k0 < Kp; k0 += 32) {
        // ---- batched loads: 4x float4 of A + 1x uint4 (8 halves) of B ----
        float4 ta[4];
#pragma unroll
        for (int j = 0; j < 4; ++j) {
            int f = tid + 256 * j;               // 0..1023 float4 slots
            int row = f >> 3;
            int kq = (f & 7) * 4;
            ta[j] = *(const float4*)(X + (size_t)(m0 + row) * Kp + k0 + kq);
        }
        const uint4 tb = *(const uint4*)(Wp + (size_t)(n0 + bn) * Kp + k0 + bkh);
        // ---- convert + store to LDS ----
#pragma unroll
        for (int j = 0; j < 4; ++j) {
            int f = tid + 256 * j;
            int row = f >> 3;
            int kq = (f & 7) * 4;
            As[row][kq + 0] = (_Float16)ta[j].x;
            As[row][kq + 1] = (_Float16)ta[j].y;
            As[row][kq + 2] = (_Float16)ta[j].z;
            As[row][kq + 3] = (_Float16)ta[j].w;
        }
        *((uint4*)&Bs[bn][bkh]) = tb;
        // prefetch next A K-tile into cache (global_prefetch_b8)
        if (k0 + 32 < Kp)
            __builtin_prefetch(X + (size_t)(m0 + (tid >> 1)) * Kp + k0 + 32 + (tid & 1) * 16, 0, 3);
        __syncthreads();

        // ---- WMMA fragments (ISA 16-bit A 16x32 / B 32x16 layouts) ----
        v16h af[2], bf[2];
        const int ar0 = wm * 32 + l15;
        const int bc0 = wn * 32 + l15;
#pragma unroll
        for (int h = 0; h < 16; ++h) {
            int ka = (h < 8 ? h : h + 8) + hsel * 8;
            af[0][h] = As[ar0][ka];
            af[1][h] = As[ar0 + 16][ka];
            int kb = (h < 8 ? h : h + 8) + hsel * 8;
            bf[0][h] = Bs[bc0][kb];
            bf[1][h] = Bs[bc0 + 16][kb];
        }
#pragma unroll
        for (int i2 = 0; i2 < 2; ++i2)
#pragma unroll
            for (int j2 = 0; j2 < 2; ++j2)
                acc[i2][j2] = __builtin_amdgcn_wmma_f32_16x16x32_f16(
                    false, af[i2], false, bf[j2], (short)0, acc[i2][j2], false, false);
        __syncthreads();
    }
    // ---- epilogue: y = acc*scale + shift, optional relu ----
#pragma unroll
    for (int i2 = 0; i2 < 2; ++i2) {
#pragma unroll
        for (int j2 = 0; j2 < 2; ++j2) {
            int gc = n0 + wn * 32 + j2 * 16 + l15;
            if (gc < ldY) {
                float s = sc[gc], t = sh[gc];
                int rbase = m0 + wm * 32 + i2 * 16 + hsel * 8;
#pragma unroll
                for (int r = 0; r < 8; ++r) {
                    float y = acc[i2][j2][r] * s + t;
                    if (relu) y = fmaxf(y, 0.0f);
                    Y[(size_t)(rbase + r) * ldY + gc] = y;
                }
            }
        }
    }
}

// ---------------------------------------------------------------------------
// Pack W[Co,Ci] fp32 -> Wp[CoP,KiP] fp16 with zero padding.
// ---------------------------------------------------------------------------
__global__ void packw_kernel(const float* __restrict__ W, _Float16* __restrict__ Wp,
                             int Co, int Ci, int CoP, int KiP)
{
    int i = blockIdx.x * blockDim.x + threadIdx.x;
    if (i >= CoP * KiP) return;
    int c = i % KiP;
    int r = i / KiP;
    Wp[i] = (_Float16)((r < Co && c < Ci) ? W[(size_t)r * Ci + c] : 0.0f);
}

// ---------------------------------------------------------------------------
// Fold bias + batchnorm into per-channel scale/shift (zero padded to CoP).
// cbr: y = ((x@W^T + b) - m) * g/sqrt(v+eps) + beta
// ---------------------------------------------------------------------------
__global__ void prep_kernel(const float* __restrict__ b, const float* __restrict__ beta,
                            const float* __restrict__ g, const float* __restrict__ m,
                            const float* __restrict__ v,
                            float* __restrict__ scale, float* __restrict__ shift,
                            int Co, int CoP, int hasbn)
{
    int i = blockIdx.x * blockDim.x + threadIdx.x;
    if (i >= CoP) return;
    if (i >= Co) { scale[i] = 0.0f; shift[i] = 0.0f; return; }
    float s = 1.0f;
    float t = b ? b[i] : 0.0f;
    if (hasbn) {
        s = g[i] * rsqrtf(v[i] + 1e-5f);
        t = (t - m[i]) * s + beta[i];
    }
    scale[i] = s;
    shift[i] = t;
}

// ---------------------------------------------------------------------------
// [B, C, N] -> channel-last copies: l0p = [B,N,C], l0xyz = [B,N,3]
// ---------------------------------------------------------------------------
__global__ void chlast_kernel(const float* __restrict__ xin,
                              float* __restrict__ l0xyz, float* __restrict__ l0p,
                              int B_, int C, int N_)
{
    int i = blockIdx.x * blockDim.x + threadIdx.x;
    int total = B_ * C * N_;
    if (i >= total) return;
    int n = i % N_;
    int c = (i / N_) % C;
    int b = i / (N_ * C);
    float v = xin[i];
    l0p[((size_t)b * N_ + n) * C + c] = v;
    if (c < 3) l0xyz[((size_t)b * N_ + n) * 3 + c] = v;
}

// xyz [rows,3] -> padded [rows,32] with zero pad cols
__global__ void pad3_kernel(const float* __restrict__ in, float* __restrict__ out, int rows)
{
    int i = blockIdx.x * blockDim.x + threadIdx.x;
    if (i >= rows * 32) return;
    int c = i % 32;
    int r = i / 32;
    out[i] = (c < 3) ? in[r * 3 + c] : 0.0f;
}

// ---------------------------------------------------------------------------
// Farthest point sampling: one block per batch, iterative argmax.
// ---------------------------------------------------------------------------
__global__ void fps_kernel(const float* __restrict__ xyz, float* __restrict__ dist,
                           int* __restrict__ out, int N_, int npoint)
{
    int b = blockIdx.x;
    const float* p = xyz + (size_t)b * N_ * 3;
    float* d = dist + (size_t)b * N_;
    int tid = threadIdx.x;
    __shared__ float smax[256];
    __shared__ int   simax[256];
    __shared__ int   sfar;
    for (int i = tid; i < N_; i += 256) d[i] = 1e10f;
    if (tid == 0) sfar = 0;
    __syncthreads();
    for (int it = 0; it < npoint; ++it) {
        int far = sfar;
        if (tid == 0) out[b * npoint + it] = far;
        float cx = p[far * 3], cy = p[far * 3 + 1], cz = p[far * 3 + 2];
        float bm = -1.0f; int bi = 0;
        for (int i = tid; i < N_; i += 256) {
            float dx = p[i * 3] - cx, dy = p[i * 3 + 1] - cy, dz = p[i * 3 + 2] - cz;
            float dd = dx * dx + dy * dy + dz * dz;
            float cur = d[i];
            if (dd < cur) { cur = dd; d[i] = dd; }
            if (cur > bm) { bm = cur; bi = i; }
        }
        smax[tid] = bm; simax[tid] = bi;
        __syncthreads();
        for (int s = 128; s > 0; s >>= 1) {
            if (tid < s && smax[tid + s] > smax[tid]) {
                smax[tid] = smax[tid + s]; simax[tid] = simax[tid + s];
            }
            __syncthreads();
        }
        if (tid == 0) sfar = simax[0];
        __syncthreads();
    }
}

__global__ void gather3_kernel(const float* __restrict__ xyz, const int* __restrict__ idx,
                               float* __restrict__ out, int Nin, int M, int B_)
{
    int i = blockIdx.x * blockDim.x + threadIdx.x;
    if (i >= B_ * M) return;
    int b = i / M;
    int j = idx[i];
    const float* s = xyz + ((size_t)b * Nin + j) * 3;
    out[i * 3 + 0] = s[0];
    out[i * 3 + 1] = s[1];
    out[i * 3 + 2] = s[2];
}

// ---------------------------------------------------------------------------
// Ball query: first K neighbors (in index order) within radius; pad with first.
// ---------------------------------------------------------------------------
__global__ void ballq_kernel(const float* __restrict__ xyz, const float* __restrict__ nxyz,
                             int* __restrict__ out, int B_, int N_, int M, int K, float r2)
{
    int i = blockIdx.x * blockDim.x + threadIdx.x;
    if (i >= B_ * M) return;
    int b = i / M;
    const float* p = xyz + (size_t)b * N_ * 3;
    float qx = nxyz[i * 3], qy = nxyz[i * 3 + 1], qz = nxyz[i * 3 + 2];
    int* o = out + (size_t)i * K;
    int cnt = 0;
    for (int j = 0; j < N_ && cnt < K; ++j) {
        float dx = p[j * 3] - qx, dy = p[j * 3 + 1] - qy, dz = p[j * 3 + 2] - qz;
        if (dx * dx + dy * dy + dz * dz <= r2) o[cnt++] = j;
    }
    int f = (cnt > 0) ? o[0] : 0;
    for (int k = cnt; k < K; ++k) o[k] = f;
}

// ---------------------------------------------------------------------------
// Grouping: out[(b,m,k), :] = concat(points[b,idx], xyz[b,idx]-new_xyz[b,m]),
// row stride GS, zero-padded columns.
// ---------------------------------------------------------------------------
__global__ void group_kernel(const float* __restrict__ pts, const float* __restrict__ xyz,
                             const float* __restrict__ nxyz, const int* __restrict__ idx,
                             float* __restrict__ out, int B_, int N_, int M, int K,
                             int Cp, int GS)
{
    int i = blockIdx.x * blockDim.x + threadIdx.x;
    if (i >= B_ * M * K) return;
    int k = i % K;
    int m = (i / K) % M;
    int b = i / (K * M);
    int j = idx[((size_t)(b * M + m)) * K + k];
    const float* src = pts + ((size_t)b * N_ + j) * Cp;
    float* dst = out + (size_t)i * GS;
    for (int c = 0; c < Cp; ++c) dst[c] = src[c];
    const float* px = xyz + ((size_t)b * N_ + j) * 3;
    const float* nx = nxyz + ((size_t)(b * M + m)) * 3;
    dst[Cp + 0] = px[0] - nx[0];
    dst[Cp + 1] = px[1] - nx[1];
    dst[Cp + 2] = px[2] - nx[2];
    for (int c = Cp + 3; c < GS; ++c) dst[c] = 0.0f;
}

__global__ void maxpool_kernel(const float* __restrict__ in, float* __restrict__ out,
                               int RM, int K, int C, int ldO, int colOff)
{
    int i = blockIdx.x * blockDim.x + threadIdx.x;
    if (i >= RM * C) return;
    int c = i % C;
    int rm = i / C;
    const float* p = in + ((size_t)rm * K) * C + c;
    float mx = -1e30f;
    for (int k = 0; k < K; ++k) mx = fmaxf(mx, p[(size_t)k * C]);
    out[(size_t)rm * ldO + colOff + c] = mx;
}

__global__ void copycols_kernel(const float* __restrict__ in, float* __restrict__ out,
                                int Rows, int C, int ldO, int colOff)
{
    int i = blockIdx.x * blockDim.x + threadIdx.x;
    if (i >= Rows * C) return;
    int c = i % C;
    int r = i / C;
    out[(size_t)r * ldO + colOff + c] = in[(size_t)r * C + c];
}

// ---------------------------------------------------------------------------
// 3-NN inverse-distance interpolation (feature propagation).
// ---------------------------------------------------------------------------
__global__ void interp_kernel(const float* __restrict__ x1, const float* __restrict__ x2,
                              const float* __restrict__ p2, float* __restrict__ out,
                              int B_, int N1, int N2, int C2, int ldO, int colOff)
{
    int i = blockIdx.x * blockDim.x + threadIdx.x;
    if (i >= B_ * N1) return;
    int b = i / N1;
    const float* q = x1 + (size_t)i * 3;
    const float* s = x2 + (size_t)b * N2 * 3;
    float d0 = 1e30f, d1 = 1e30f, d2 = 1e30f;
    int i0 = 0, i1 = 0, i2 = 0;
    for (int j = 0; j < N2; ++j) {
        float dx = s[j * 3] - q[0], dy = s[j * 3 + 1] - q[1], dz = s[j * 3 + 2] - q[2];
        float dd = dx * dx + dy * dy + dz * dz;
        if (dd < d0)      { d2 = d1; i2 = i1; d1 = d0; i1 = i0; d0 = dd; i0 = j; }
        else if (dd < d1) { d2 = d1; i2 = i1; d1 = dd; i1 = j; }
        else if (dd < d2) { d2 = dd; i2 = j; }
    }
    float w0 = 1.0f / (d0 + 1e-8f), w1 = 1.0f / (d1 + 1e-8f), w2 = 1.0f / (d2 + 1e-8f);
    float ws = w0 + w1 + w2;
    w0 /= ws; w1 /= ws; w2 /= ws;
    const float* a  = p2 + ((size_t)b * N2 + i0) * C2;
    const float* bb = p2 + ((size_t)b * N2 + i1) * C2;
    const float* cc = p2 + ((size_t)b * N2 + i2) * C2;
    float* o = out + (size_t)i * ldO + colOff;
    for (int c = 0; c < C2; ++c) o[c] = w0 * a[c] + w1 * bb[c] + w2 * cc[c];
}

// ---------------------------------------------------------------------------
// LayerNorm over last dim (D=128): one 128-thread block per row.
// ---------------------------------------------------------------------------
__global__ void ln_kernel(const float* __restrict__ in, float* __restrict__ out,
                          const float* __restrict__ g, const float* __restrict__ b, int D_)
{
    int row = blockIdx.x;
    int t = threadIdx.x;
    const float* x = in + (size_t)row * D_;
    __shared__ float s1[128], s2[128];
    float v = x[t];
    s1[t] = v; s2[t] = v * v;
    __syncthreads();
    for (int s = 64; s > 0; s >>= 1) {
        if (t < s) { s1[t] += s1[t + s]; s2[t] += s2[t + s]; }
        __syncthreads();
    }
    float mu = s1[0] / D_;
    float var = s2[0] / D_ - mu * mu;
    float r = rsqrtf(var + 1e-5f);
    out[(size_t)row * D_ + t] = (v - mu) * r * g[t] + b[t];
}

__global__ void add_kernel(const float* __restrict__ a, const float* __restrict__ b,
                           float* __restrict__ o, int n)
{
    int i = blockIdx.x * blockDim.x + threadIdx.x;
    if (i < n) o[i] = a[i] + b[i];
}

// ---------------------------------------------------------------------------
// Flash attention, one wave per query row. hd==32 -> one lane per head dim.
// ---------------------------------------------------------------------------
__global__ void attn_kernel(const float* __restrict__ qkv, float* __restrict__ out,
                            int B_, int S_, int H_, int hd)
{
    int gid = blockIdx.x * blockDim.x + threadIdx.x;
    int wid = gid >> 5;
    int lane = threadIdx.x & 31;
    int total = B_ * H_ * S_;
    if (wid >= total) return;
    int qi = wid % S_;
    int h = (wid / S_) % H_;
    int b = wid / (S_ * H_);
    int D_ = H_ * hd;
    const float* base = qkv + (size_t)b * S_ * 3 * D_;
    float qv = base[(size_t)qi * 3 * D_ + h * hd + lane] * rsqrtf((float)hd);
    float mx = -1e30f, ls = 0.0f, ac = 0.0f;
    for (int j = 0; j < S_; ++j) {
        const float* row = base + (size_t)j * 3 * D_;
        float d = qv * row[D_ + h * hd + lane];
#pragma unroll
        for (int o = 16; o > 0; o >>= 1) d += __shfl_xor(d, o, 32);
        float nm = fmaxf(mx, d);
        float scn = expf(mx - nm);
        float e = expf(d - nm);
        ls = ls * scn + e;
        ac = ac * scn + e * row[2 * D_ + h * hd + lane];
        mx = nm;
    }
    out[(size_t)(b * S_ + qi) * D_ + h * hd + lane] = ac / ls;
}

// ---------------------------------------------------------------------------
// Final: pack heads channel-first; normalize direction. Head buffers stride 32.
// ---------------------------------------------------------------------------
__global__ void final_kernel(const float* __restrict__ path, const float* __restrict__ keyp,
                             const float* __restrict__ dirn, float* __restrict__ out,
                             int B_, int N_)
{
    int i = blockIdx.x * blockDim.x + threadIdx.x;
    if (i >= B_ * N_) return;
    int b = i / N_;
    int n = i % N_;
    out[i] = path[(size_t)i * 32];
    out[(size_t)B_ * N_ + i] = keyp[(size_t)i * 32];
    float dx = dirn[(size_t)i * 32], dy = dirn[(size_t)i * 32 + 1], dz = dirn[(size_t)i * 32 + 2];
    float nn = fmaxf(sqrtf(dx * dx + dy * dy + dz * dz), 1e-12f);
    size_t base = (size_t)2 * B_ * N_;
    out[base + ((size_t)b * 3 + 0) * N_ + n] = dx / nn;
    out[base + ((size_t)b * 3 + 1) * N_ + n] = dy / nn;
    out[base + ((size_t)b * 3 + 2) * N_ + n] = dz / nn;
}

// ============================================================================
// Host orchestration. d_in: 0=xyz, then params leaves (sorted-dict pytree).
// ============================================================================
extern "C" void kernel_launch(void* const* d_in, const int* in_sizes, int n_in,
                              void* d_out, int out_size, void* d_ws, size_t ws_size,
                              hipStream_t stream)
{
    (void)in_sizes; (void)n_in; (void)out_size; (void)ws_size;
    const int B = 2, N = 4096, S = 4096, H = 4, HD = 32;
    const int Srows = B * N;

    auto F = [&](int i) -> const float* { return (const float*)d_in[i]; };
    auto pad32 = [](int x) { return (x + 31) & ~31; };
    auto pad64 = [](int x) { return (x + 63) & ~63; };

    char* wsp = (char*)d_ws;
    size_t off = 0;
    auto alloc = [&](size_t nf) -> float* {
        float* p = (float*)(wsp + off);
        off += ((nf * sizeof(float) + 255) & ~(size_t)255);
        return p;
    };

    // ---- workspace buffers ----
    float* l0_xyz  = alloc((size_t)B * N * 3);
    float* l0_xyzp = alloc((size_t)Srows * 32);     // padded for Ci=3 GEMM
    float* l0_p    = alloc((size_t)B * N * 6);
    float* fpsd    = alloc((size_t)B * N);
    int*   fidx    = (int*)alloc((size_t)B * 1024);
    int*   bqidx   = (int*)alloc((size_t)B * 1024 * 32);
    float* gbufA   = alloc((size_t)8192 * 2048);
    float* gbufB   = alloc((size_t)8192 * 2048);
    float* l1_xyz  = alloc((size_t)B * 1024 * 3);
    float* l1_p    = alloc((size_t)B * 1024 * 96);
    float* l2_xyz  = alloc((size_t)B * 256 * 3);
    float* l2_p    = alloc((size_t)B * 256 * 256);
    float* l3_xyz  = alloc((size_t)B * 64 * 3);
    float* l3_p    = alloc((size_t)B * 64 * 512);
    float* l4_xyz  = alloc((size_t)B * 16 * 3);
    float* l4_p    = alloc((size_t)B * 16 * 1024);
    float* l3_pn   = alloc((size_t)B * 64 * 256);
    float* l2_pn   = alloc((size_t)B * 256 * 256);
    float* l1_pn   = alloc((size_t)B * 1024 * 128);
    float* l0_f    = alloc((size_t)Srows * 128);
    float* posb    = alloc((size_t)Srows * 128);
    float* xbuf    = alloc((size_t)Srows * 128);
    float* x0      = alloc((size_t)Srows * 128);
    float* xfin    = alloc((size_t)Srows * 128);
    float* t1      = alloc((size_t)Srows * 128);
    float* t2      = alloc((size_t)Srows * 128);
    float* qkvb    = alloc((size_t)Srows * 384);
    float* attno   = alloc((size_t)Srows * 128);
    float* hbuf    = alloc((size_t)Srows * 128);
    float* pathb   = alloc((size_t)Srows * 32);
    float* keypb   = alloc((size_t)Srows * 32);
    float* dirnb   = alloc((size_t)Srows * 32);

    // ---- packed-GEMM helpers ----
    struct PW { const _Float16* Wp; const float* sc; const float* sh; int KiP; int CoP; };
    auto packw = [&](const float* W, const float* b, const float* beta, const float* g,
                     const float* m, const float* v, int Co, int Ci, int hasbn) -> PW {
        PW p;
        p.KiP = pad32(Ci);
        p.CoP = pad64(Co);
        _Float16* wp = (_Float16*)alloc((size_t)p.CoP * p.KiP / 2);
        float* sc = alloc((size_t)p.CoP);
        float* sh = alloc((size_t)p.CoP);
        packw_kernel<<<(p.CoP * p.KiP + TPB - 1) / TPB, TPB, 0, stream>>>(W, wp, Co, Ci, p.CoP, p.KiP);
        prep_kernel<<<(p.CoP + TPB - 1) / TPB, TPB, 0, stream>>>(b, beta, g, m, v, sc, sh, Co, p.CoP, hasbn);
        p.Wp = wp; p.sc = sc; p.sh = sh;
        return p;
    };
    auto gemm = [&](const float* X, const PW& p, float* Y, int R, int ldY, int relu) {
        dim3 grid(R / 128, p.CoP / 64);
        gemm2_kernel<<<grid, 256, 0, stream>>>(X, p.Wp, p.sc, p.sh, Y, p.KiP, ldY, relu);
    };
    auto cbr = [&](const float* X, float* Y, int R, int Ci, int Co, int pi) {
        // cbn leaf order: W, b, beta, g, m, v
        PW p = packw(F(pi), F(pi + 1), F(pi + 2), F(pi + 3), F(pi + 4), F(pi + 5), Co, Ci, 1);
        gemm(X, p, Y, R, pad32(Co), 1);
    };
    auto linear = [&](const float* X, const float* W, const float* bias, float* Y,
                      int R, int Ci, int Co, int relu) {
        PW p = packw(W, bias, nullptr, nullptr, nullptr, nullptr, Co, Ci, 0);
        gemm(X, p, Y, R, pad32(Co), relu);
    };

    // ---- set abstraction (MSG) level ----
    auto run_sa = [&](const float* xyz_in, const float* pts, int Nin, int Cp, int M,
                      float* oxyz, float* op, int Ctot,
                      int base0, const int* ch0, float r0, int k0,
                      int base1, const int* ch1, float r1, int k1) {
        fps_kernel<<<B, 256, 0, stream>>>(xyz_in, fpsd, fidx, Nin, M);
        gather3_kernel<<<(B * M + TPB - 1) / TPB, TPB, 0, stream>>>(xyz_in, fidx, oxyz, Nin, M, B);
        int bases[2] = {base0, base1};
        const int* chs[2] = {ch0, ch1};
        float rr[2] = {r0, r1};
        int ks[2] = {k0, k1};
        int colOff = 0;
        for (int br = 0; br < 2; ++br) {
            int K = ks[br];
            float r = rr[br];
            int rows = B * M * K;                 // always multiple of 128
            int GS = pad32(Cp + 3);
            ballq_kernel<<<(B * M + TPB - 1) / TPB, TPB, 0, stream>>>(
                xyz_in, oxyz, bqidx, B, Nin, M, K, r * r);
            group_kernel<<<(rows + TPB - 1) / TPB, TPB, 0, stream>>>(
                pts, xyz_in, oxyz, bqidx, gbufA, B, Nin, M, K, Cp, GS);
            const int* ch = chs[br];
            int pb = bases[br];
            cbr(gbufA, gbufB, rows, Cp + 3, ch[0], pb);
            cbr(gbufB, gbufA, rows, ch[0], ch[1], pb + 6);
            cbr(gbufA, gbufB, rows, ch[1], ch[2], pb + 12);
            maxpool_kernel<<<(B * M * ch[2] + TPB - 1) / TPB, TPB, 0, stream>>>(
                gbufB, op, B * M, K, ch[2], Ctot, colOff);
            colOff += ch[2];
        }
    };

    // ---- feature propagation level ----
    auto run_fp = [&](const float* x1, const float* x2, const float* p1, int C1,
                      const float* p2, int C2, int N1, int N2,
                      int pb, int c0, int c1, float* out) {
        int Cin = C1 + C2;                        // always multiple of 32
        int rows = B * N1;                        // always multiple of 128
        if (p1)
            copycols_kernel<<<(rows * C1 + TPB - 1) / TPB, TPB, 0, stream>>>(
                p1, gbufA, rows, C1, Cin, 0);
        interp_kernel<<<(rows + TPB - 1) / TPB, TPB, 0, stream>>>(
            x1, x2, p2, gbufA, B, N1, N2, C2, Cin, C1);
        cbr(gbufA, gbufB, rows, Cin, c0, pb);
        cbr(gbufB, out, rows, c0, c1, pb + 6);
    };

    // ===== pipeline =====
    chlast_kernel<<<(B * 6 * N + TPB - 1) / TPB, TPB, 0, stream>>>(F(0), l0_xyz, l0_p, B, 6, N);
    pad3_kernel<<<(Srows * 32 + TPB - 1) / TPB, TPB, 0, stream>>>(l0_xyz, l0_xyzp, Srows);

    const int c10[3] = {16, 16, 32},    c11[3] = {32, 32, 64};
    const int c20[3] = {64, 64, 128},   c21[3] = {64, 96, 128};
    const int c30[3] = {128, 196, 256}, c31[3] = {128, 196, 256};
    const int c40[3] = {256, 256, 512}, c41[3] = {256, 384, 512};

    run_sa(l0_xyz, l0_p, N,    6,   1024, l1_xyz, l1_p, 96,   77,  c10, 0.05f, 16, 95,  c11, 0.1f, 32);
    run_sa(l1_xyz, l1_p, 1024, 96,  256,  l2_xyz, l2_p, 256,  113, c20, 0.1f,  16, 131, c21, 0.2f, 32);
    run_sa(l2_xyz, l2_p, 256,  256, 64,   l3_xyz, l3_p, 512,  149, c30, 0.2f,  16, 167, c31, 0.4f, 32);
    run_sa(l3_xyz, l3_p, 64,   512, 16,   l4_xyz, l4_p, 1024, 185, c40, 0.4f,  16, 203, c41, 0.8f, 32);

    run_fp(l3_xyz, l4_xyz, l3_p,    512, l4_p,  1024, 64,   16,   45, 256, 256, l3_pn);
    run_fp(l2_xyz, l3_xyz, l2_p,    256, l3_pn, 256,  256,  64,   33, 256, 256, l2_pn);
    run_fp(l1_xyz, l2_xyz, l1_p,    96,  l2_pn, 256,  1024, 256,  21, 256, 128, l1_pn);
    run_fp(l0_xyz, l1_xyz, nullptr, 0,   l1_pn, 128,  4096, 1024, 9,  128, 128, l0_f);

    // positional encoding: relu(l0_xyz @ W1^T + b1) @ W2^T + b2
    linear(l0_xyzp, F(73), F(75), t1, Srows, 3, 128, 1);
    linear(t1, F(74), F(76), posb, Srows, 128, 128, 0);
    add_kernel<<<(Srows * 128 + TPB - 1) / TPB, TPB, 0, stream>>>(l0_f, posb, xbuf, Srows * 128);
    copycols_kernel<<<(Srows * 128 + TPB - 1) / TPB, TPB, 0, stream>>>(xbuf, x0, Srows, 128, 128, 0);

    // ---- 2 transformer encoder layers ----
    // enc leaves: W1,W2,Wo,Wqkv,b1,b2,bo,bqkv,ln1b,ln1g,ln2b,ln2g
    int encBase[2] = {221, 233};
    int nwaves = B * H * S;
    for (int L = 0; L < 2; ++L) {
        int e = encBase[L];
        linear(xbuf, F(e + 3), F(e + 7), qkvb, Srows, 128, 384, 0);       // qkv
        attn_kernel<<<(nwaves * 32 + TPB - 1) / TPB, TPB, 0, stream>>>(qkvb, attno, B, S, H, HD);
        linear(attno, F(e + 2), F(e + 6), t1, Srows, 128, 128, 0);        // Wo
        add_kernel<<<(Srows * 128 + TPB - 1) / TPB, TPB, 0, stream>>>(xbuf, t1, t2, Srows * 128);
        ln_kernel<<<Srows, 128, 0, stream>>>(t2, xbuf, F(e + 9), F(e + 8), 128);   // ln1
        linear(xbuf, F(e + 0), F(e + 4), gbufA, Srows, 128, 2048, 1);     // FFN up + relu
        linear(gbufA, F(e + 1), F(e + 5), t1, Srows, 2048, 128, 0);       // FFN down
        add_kernel<<<(Srows * 128 + TPB - 1) / TPB, TPB, 0, stream>>>(xbuf, t1, t2, Srows * 128);
        ln_kernel<<<Srows, 128, 0, stream>>>(t2, xbuf, F(e + 11), F(e + 10), 128); // ln2
    }
    // transformer output MLP + outer residual
    linear(xbuf, F(245), F(247), t1, Srows, 128, 128, 1);      // mW1, mb1 (+relu)
    linear(t1, F(246), F(248), t2, Srows, 128, 128, 0);        // mW2, mb2
    add_kernel<<<(Srows * 128 + TPB - 1) / TPB, TPB, 0, stream>>>(t2, x0, xfin, Srows * 128);

    // ---- heads (leaf order: W2, b2, l1-cbn) ----
    cbr(xfin, hbuf, Srows, 128, 128, 67);                      // path_head.l1
    linear(hbuf, F(65), F(66), pathb, Srows, 128, 1, 0);
    cbr(xfin, hbuf, Srows, 128, 128, 59);                      // keypoint_head.l1
    linear(hbuf, F(57), F(58), keypb, Srows, 128, 1, 0);
    cbr(xfin, hbuf, Srows, 128, 128, 3);                       // direction_head.l1
    linear(hbuf, F(1), F(2), dirnb, Srows, 128, 3, 0);

    final_kernel<<<(Srows + TPB - 1) / TPB, TPB, 0, stream>>>(pathb, keypb, dirnb,
                                                              (float*)d_out, B, N);
}
